// DeepSeekMoE_43619687858993
// MI455X (gfx1250) — compile-verified, matrix operand order checked by reading, not run on hardware
//
#include <hip/hip_runtime.h>
#include <math.h>

#define HIDDEN 1024
#define INTER  2048
#define NE     8
#define NTOK   8192          // 4 * 2048 tokens
#define BM     64
#define BN     64
#define BK     32
#define LDT    40            // padded LDS row stride (bf16 elems), 80B

typedef __attribute__((ext_vector_type(16))) __bf16 v16bf;
typedef __attribute__((ext_vector_type(8)))  __bf16 v8bf;
typedef __attribute__((ext_vector_type(8)))  float  v8f;

static __device__ __forceinline__ __bf16 f2bf(float f) {
    unsigned u; __builtin_memcpy(&u, &f, 4);
    u = (u + 0x7FFFu + ((u >> 16) & 1u)) >> 16;        // round-to-nearest-even
    unsigned short s = (unsigned short)u;
    __bf16 b; __builtin_memcpy(&b, &s, 2);
    return b;
}

// Async 16B global -> LDS copy (CDNA5 GLOBAL_LOAD_ASYNC_TO_LDS_B128, ASYNCcnt-tracked).
// LDS address = low 32 bits of the generic pointer (ISA: LDS aperture uses addr[31:0]).
static __device__ __forceinline__ void async_cp16(void* lds, const void* gptr) {
    unsigned l = (unsigned)(size_t)lds;
    asm volatile("global_load_async_to_lds_b128 %0, %1, off"
                 :: "v"(l), "v"(gptr) : "memory");
}
#define WAIT_ASYNC(n) asm volatile("s_wait_asynccnt %0" :: "i"(n) : "memory")

// Load one 16x32 bf16 WMMA fragment from an LDS tile (row stride LDT).
// Lane L: row = L%16, K-chunks at (L/16)*8 and 16+(L/16)*8 (ISA 7.12.2).
static __device__ __forceinline__ v16bf ldfrag(const __bf16* base, int lane) {
    const __bf16* p = base + (lane & 15) * LDT + ((lane >> 4) << 3);
    v8bf lo = *(const v8bf*)(p);
    v8bf hi = *(const v8bf*)(p + 16);
    v16bf r;
#pragma unroll
    for (int i = 0; i < 8; ++i) { r[i] = lo[i]; r[8 + i] = hi[i]; }
    return r;
}

// ---------------------------------------------------------------- utilities
__global__ void dsmoe_zero_cnt(int* cnt) {
    if (threadIdx.x < NE) cnt[threadIdx.x] = 0;
}

__global__ __launch_bounds__(256)
void dsmoe_cast_w(const float* __restrict__ src, __bf16* __restrict__ dst, int n4) {
    int i = blockIdx.x * 256 + threadIdx.x;
    if (i < n4) {
        float4 v = ((const float4*)src)[i];
        __bf16 b[4] = { f2bf(v.x), f2bf(v.y), f2bf(v.z), f2bf(v.w) };
        unsigned long long pk; __builtin_memcpy(&pk, b, 8);
        ((unsigned long long*)dst)[i] = pk;
    }
}

// ---------------------------------------------------------------- router
__global__ __launch_bounds__(256)
void dsmoe_router(const float* __restrict__ x, const float* __restrict__ wr,
                  __bf16* __restrict__ xb, int* __restrict__ cnt, int* __restrict__ list) {
    __shared__ float xs[HIDDEN];
    __shared__ float lg[NE];
    const int t   = blockIdx.x;
    const int tid = threadIdx.x;
    const float* xrow  = x  + (size_t)t * HIDDEN;
    __bf16*      xbrow = xb + (size_t)t * HIDDEN;

    for (int i = tid; i < HIDDEN / 4; i += 256) {
        float4 v = ((const float4*)xrow)[i];
        xs[4*i+0] = v.x; xs[4*i+1] = v.y; xs[4*i+2] = v.z; xs[4*i+3] = v.w;
        __bf16 b[4] = { f2bf(v.x), f2bf(v.y), f2bf(v.z), f2bf(v.w) };
        unsigned long long pk; __builtin_memcpy(&pk, b, 8);
        *(unsigned long long*)(xbrow + 4*i) = pk;
    }
    __syncthreads();

    const int w = tid >> 5, lane = tid & 31;   // 8 waves, one expert each (wave32)
    const float* wrow = wr + w * HIDDEN;
    float s = 0.f;
    for (int j = lane; j < HIDDEN; j += 32) s = fmaf(xs[j], wrow[j], s);
#pragma unroll
    for (int off = 16; off > 0; off >>= 1) s += __shfl_down(s, off, 32);
    if (lane == 0) lg[w] = s;
    __syncthreads();

    if (tid == 0) {
        int e0 = 0; float b0 = lg[0];
#pragma unroll
        for (int e = 1; e < NE; ++e) if (lg[e] > b0) { b0 = lg[e]; e0 = e; }
        int e1 = -1; float b1 = -3.4e38f;
#pragma unroll
        for (int e = 0; e < NE; ++e) if (e != e0 && lg[e] > b1) { b1 = lg[e]; e1 = e; }
        int p0 = atomicAdd(&cnt[e0], 1); list[e0 * NTOK + p0] = (t << 1);
        int p1 = atomicAdd(&cnt[e1], 1); list[e1 * NTOK + p1] = (t << 1) | 1;
    }
}

// ---------------------------------------------------------------- GEMM1: gate/up + SiLU, gathered rows
__global__ __launch_bounds__(128)
void dsmoe_gemm1(const __bf16* __restrict__ xb, const __bf16* __restrict__ wgb,
                 const __bf16* __restrict__ wub, __bf16* __restrict__ h,
                 const int* __restrict__ cnt, const int* __restrict__ list) {
    const int e  = blockIdx.z;
    const int m0 = blockIdx.y * BM;
    const int n0 = blockIdx.x * BN;
    const int cntE = cnt[e];
    if (m0 >= cntE) return;

    __shared__ __bf16 As [2][BM * LDT];
    __shared__ __bf16 Bgs[2][BN * LDT];
    __shared__ __bf16 Bus[2][BN * LDT];
    __shared__ int    lvs[BM];

    const int tid  = threadIdx.x;
    const int lane = tid & 31;
    const int wid  = tid >> 5;
    const int wm   = wid >> 1, wn = wid & 1;

    const int r0 = tid >> 2;     // staging row 0..31 (and +32)
    const int c0 = tid & 3;      // 16B chunk within the 32-elem K tile
    int lv0 = -1, lv1 = -1;
    if (m0 + r0      < cntE) lv0 = list[e * NTOK + m0 + r0];
    if (m0 + r0 + 32 < cntE) lv1 = list[e * NTOK + m0 + r0 + 32];
    if (c0 == 0) { lvs[r0] = lv0; lvs[r0 + 32] = lv1; }

    // Padded rows redirect to token 0: garbage in -> rows whose results are never stored.
    // Keeps EXEC uniform (WMMA requirement) and the async pipeline branch-free.
    const size_t wbase = (size_t)e * INTER * HIDDEN;
    const __bf16* wg0 = wgb + wbase + (size_t)(n0 + r0)      * HIDDEN + c0 * 8;
    const __bf16* wg1 = wgb + wbase + (size_t)(n0 + r0 + 32) * HIDDEN + c0 * 8;
    const __bf16* wu0 = wub + wbase + (size_t)(n0 + r0)      * HIDDEN + c0 * 8;
    const __bf16* wu1 = wub + wbase + (size_t)(n0 + r0 + 32) * HIDDEN + c0 * 8;
    const __bf16* xa0 = xb + (size_t)((lv0 >= 0 ? lv0 : 0) >> 1) * HIDDEN + c0 * 8;
    const __bf16* xa1 = xb + (size_t)((lv1 >= 0 ? lv1 : 0) >> 1) * HIDDEN + c0 * 8;

    const int sA0 =  r0       * LDT + c0 * 8;
    const int sA1 = (r0 + 32) * LDT + c0 * 8;

    v8f cg[2][2] = {}; v8f cu[2][2] = {};

    // ---- async double-buffered pipeline: 6 ASYNC b128 per stage per wave
    {   // prologue: stage 0 into buffer 0
        async_cp16(&As [0][sA0], xa0);
        async_cp16(&As [0][sA1], xa1);
        async_cp16(&Bgs[0][sA0], wg0);
        async_cp16(&Bgs[0][sA1], wg1);
        async_cp16(&Bus[0][sA0], wu0);
        async_cp16(&Bus[0][sA1], wu1);
    }
    const int NIT = HIDDEN / BK;
    for (int it = 0; it < NIT; ++it) {
        const int kn = (it + 1) * BK;
        if (it + 1 < NIT) {
            const int b = (it + 1) & 1;
            async_cp16(&As [b][sA0], xa0 + kn);
            async_cp16(&As [b][sA1], xa1 + kn);
            async_cp16(&Bgs[b][sA0], wg0 + kn);
            async_cp16(&Bgs[b][sA1], wg1 + kn);
            async_cp16(&Bus[b][sA0], wu0 + kn);
            async_cp16(&Bus[b][sA1], wu1 + kn);
            WAIT_ASYNC(6);       // only next stage still in flight (in-order completion)
        } else {
            WAIT_ASYNC(0);
        }
        __syncthreads();         // stage `it` visible to all waves

        const int b = it & 1;
        v16bf a[2], bg[2], bu[2];
        a [0] = ldfrag(&As [b][(wm * 32     ) * LDT], lane);
        a [1] = ldfrag(&As [b][(wm * 32 + 16) * LDT], lane);
        bg[0] = ldfrag(&Bgs[b][(wn * 32     ) * LDT], lane);
        bg[1] = ldfrag(&Bgs[b][(wn * 32 + 16) * LDT], lane);
        bu[0] = ldfrag(&Bus[b][(wn * 32     ) * LDT], lane);
        bu[1] = ldfrag(&Bus[b][(wn * 32 + 16) * LDT], lane);
#pragma unroll
        for (int mi = 0; mi < 2; ++mi)
#pragma unroll
            for (int ni = 0; ni < 2; ++ni) {
                cg[mi][ni] = __builtin_amdgcn_wmma_f32_16x16x32_bf16(
                    false, a[mi], false, bg[ni], (short)0, cg[mi][ni], false, false);
                cu[mi][ni] = __builtin_amdgcn_wmma_f32_16x16x32_bf16(
                    false, a[mi], false, bu[ni], (short)0, cu[mi][ni], false, false);
            }
        __syncthreads();         // all ds reads of buffer b done before stage it+2 overwrites it
    }

    const int mb = wm * 32;
    const int nb = n0 + wn * 32;
#pragma unroll
    for (int mi = 0; mi < 2; ++mi) {
#pragma unroll
        for (int ni = 0; ni < 2; ++ni) {
            int mloc = mb + mi * 16 + ((lane >> 4) << 3);
            int ncol = nb + ni * 16 + (lane & 15);
#pragma unroll
            for (int r = 0; r < 8; ++r) {
                int lv = lvs[mloc + r];
                if (lv >= 0) {
                    float g  = cg[mi][ni][r];
                    float u  = cu[mi][ni][r];
                    float hv = (g / (1.0f + __expf(-g))) * u;   // SiLU(g)*u
                    h[(size_t)lv * INTER + ncol] = f2bf(hv);
                }
            }
        }
    }
}

// ---------------------------------------------------------------- GEMM2: h @ w_down^T -> per-slot out2
__global__ __launch_bounds__(128)
void dsmoe_gemm2(const __bf16* __restrict__ h, const __bf16* __restrict__ wdb,
                 float* __restrict__ out2, const int* __restrict__ cnt,
                 const int* __restrict__ list) {
    const int e  = blockIdx.z;
    const int m0 = blockIdx.y * BM;
    const int n0 = blockIdx.x * BN;
    const int cntE = cnt[e];
    if (m0 >= cntE) return;

    __shared__ __bf16 As [2][BM * LDT];
    __shared__ __bf16 Bds[2][BN * LDT];
    __shared__ int    lvs[BM];

    const int tid  = threadIdx.x;
    const int lane = tid & 31;
    const int wid  = tid >> 5;
    const int wm   = wid >> 1, wn = wid & 1;

    const int r0 = tid >> 2;
    const int c0 = tid & 3;
    int lv0 = -1, lv1 = -1;
    if (m0 + r0      < cntE) lv0 = list[e * NTOK + m0 + r0];
    if (m0 + r0 + 32 < cntE) lv1 = list[e * NTOK + m0 + r0 + 32];
    if (c0 == 0) { lvs[r0] = lv0; lvs[r0 + 32] = lv1; }

    const size_t wbase = (size_t)e * HIDDEN * INTER;
    const __bf16* wd0 = wdb + wbase + (size_t)(n0 + r0)      * INTER + c0 * 8;
    const __bf16* wd1 = wdb + wbase + (size_t)(n0 + r0 + 32) * INTER + c0 * 8;
    const __bf16* ha0 = h + (size_t)(lv0 >= 0 ? lv0 : 0) * INTER + c0 * 8;
    const __bf16* ha1 = h + (size_t)(lv1 >= 0 ? lv1 : 0) * INTER + c0 * 8;

    const int sA0 =  r0       * LDT + c0 * 8;
    const int sA1 = (r0 + 32) * LDT + c0 * 8;

    v8f cc[2][2] = {};

    {   // prologue: 4 ASYNC b128 per stage per wave
        async_cp16(&As [0][sA0], ha0);
        async_cp16(&As [0][sA1], ha1);
        async_cp16(&Bds[0][sA0], wd0);
        async_cp16(&Bds[0][sA1], wd1);
    }
    const int NIT = INTER / BK;
    for (int it = 0; it < NIT; ++it) {
        const int kn = (it + 1) * BK;
        if (it + 1 < NIT) {
            const int b = (it + 1) & 1;
            async_cp16(&As [b][sA0], ha0 + kn);
            async_cp16(&As [b][sA1], ha1 + kn);
            async_cp16(&Bds[b][sA0], wd0 + kn);
            async_cp16(&Bds[b][sA1], wd1 + kn);
            WAIT_ASYNC(4);
        } else {
            WAIT_ASYNC(0);
        }
        __syncthreads();

        const int b = it & 1;
        v16bf a[2], bd[2];
        a [0] = ldfrag(&As [b][(wm * 32     ) * LDT], lane);
        a [1] = ldfrag(&As [b][(wm * 32 + 16) * LDT], lane);
        bd[0] = ldfrag(&Bds[b][(wn * 32     ) * LDT], lane);
        bd[1] = ldfrag(&Bds[b][(wn * 32 + 16) * LDT], lane);
#pragma unroll
        for (int mi = 0; mi < 2; ++mi)
#pragma unroll
            for (int ni = 0; ni < 2; ++ni)
                cc[mi][ni] = __builtin_amdgcn_wmma_f32_16x16x32_bf16(
                    false, a[mi], false, bd[ni], (short)0, cc[mi][ni], false, false);
        __syncthreads();
    }

    const int mb = wm * 32;
    const int nb = n0 + wn * 32;
#pragma unroll
    for (int mi = 0; mi < 2; ++mi) {
#pragma unroll
        for (int ni = 0; ni < 2; ++ni) {
            int mloc = mb + mi * 16 + ((lane >> 4) << 3);
            int ncol = nb + ni * 16 + (lane & 15);
#pragma unroll
            for (int r = 0; r < 8; ++r) {
                int lv = lvs[mloc + r];
                if (lv >= 0)
                    out2[(size_t)lv * HIDDEN + ncol] = cc[mi][ni][r];
            }
        }
    }
}

// ---------------------------------------------------------------- finalize: out[t] = 0.25*(out2[2t]+out2[2t+1])
__global__ __launch_bounds__(256)
void dsmoe_final(const float* __restrict__ out2, float* __restrict__ out) {
    const int t = blockIdx.x, tid = threadIdx.x;       // HIDDEN/4 == 256 float4 per token
    const float4* a = (const float4*)(out2 + (size_t)(2 * t)     * HIDDEN);
    const float4* b = (const float4*)(out2 + (size_t)(2 * t + 1) * HIDDEN);
    float4* o = (float4*)(out + (size_t)t * HIDDEN);
    float4 va = a[tid], vb = b[tid], r;
    r.x = 0.25f * (va.x + vb.x);
    r.y = 0.25f * (va.y + vb.y);
    r.z = 0.25f * (va.z + vb.z);
    r.w = 0.25f * (va.w + vb.w);
    o[tid] = r;
}

// ---------------------------------------------------------------- launch
extern "C" void kernel_launch(void* const* d_in, const int* in_sizes, int n_in,
                              void* d_out, int out_size, void* d_ws, size_t ws_size,
                              hipStream_t stream) {
    (void)in_sizes; (void)n_in; (void)out_size; (void)ws_size;
    const float* x  = (const float*)d_in[0];
    const float* wr = (const float*)d_in[1];
    const float* wg = (const float*)d_in[2];
    const float* wu = (const float*)d_in[3];
    const float* wd = (const float*)d_in[4];
    float* out = (float*)d_out;

    char*  ws = (char*)d_ws;
    size_t o  = 0;
    auto alloc = [&](size_t bytes) { size_t p = o; o += (bytes + 255) & ~(size_t)255; return p; };
    __bf16* xb   = (__bf16*)(ws + alloc((size_t)NTOK * HIDDEN * 2));          //  16 MB
    __bf16* wgb  = (__bf16*)(ws + alloc((size_t)NE * INTER * HIDDEN * 2));    //  33.5 MB
    __bf16* wub  = (__bf16*)(ws + alloc((size_t)NE * INTER * HIDDEN * 2));    //  33.5 MB
    __bf16* wdb  = (__bf16*)(ws + alloc((size_t)NE * HIDDEN * INTER * 2));    //  33.5 MB
    __bf16* hbuf = (__bf16*)(ws + alloc((size_t)NTOK * 2 * INTER * 2));       //  67 MB
    float*  out2 = (float*) (ws + alloc((size_t)NTOK * 2 * HIDDEN * 4));      //  67 MB
    int*    cnt  = (int*)   (ws + alloc((size_t)NE * 4));
    int*    list = (int*)   (ws + alloc((size_t)NE * NTOK * 4));

    dsmoe_zero_cnt<<<1, 32, 0, stream>>>(cnt);

    const int n4 = NE * INTER * HIDDEN / 4;                  // float4 count per weight tensor
    dsmoe_cast_w<<<(n4 + 255) / 256, 256, 0, stream>>>(wg, wgb, n4);
    dsmoe_cast_w<<<(n4 + 255) / 256, 256, 0, stream>>>(wu, wub, n4);
    dsmoe_cast_w<<<(n4 + 255) / 256, 256, 0, stream>>>(wd, wdb, n4);

    dsmoe_router<<<NTOK, 256, 0, stream>>>(x, wr, xb, cnt, list);

    dsmoe_gemm1<<<dim3(INTER / BN, NTOK / BM, NE), 128, 0, stream>>>(xb, wgb, wub, hbuf, cnt, list);
    dsmoe_gemm2<<<dim3(HIDDEN / BN, NTOK / BM, NE), 128, 0, stream>>>(hbuf, wdb, out2, cnt, list);

    dsmoe_final<<<NTOK, 256, 0, stream>>>(out2, out);
}
